// LinearAttentionBlock_38749194944564
// MI455X (gfx1250) — matrix operand order repaired and checked by reading, not compiled, not run on hardware
//
#include <hip/hip_runtime.h>

// ---------------------------------------------------------------------------
// Linear attention block for MI455X (gfx1250), wave32, WMMA bf16.
//   B=8, L=W=64 -> N=4096 rows/batch, Mtot=32768, C=512, H=8, D=64, inner=512
// Pipeline:
//   K0: wT[n][k] = bf16(w[k][n])   (one-time 3MB->1.5MB, L2-resident)
//   K1: qkv = x @ w  (bf16 WMMA, f32 acc) + fused per-head LayerNorm(q,k)
//       A tile: fp32 load -> bf16 convert -> LDS (load-all then convert-all)
//       B tile: raw bf16 via GLOBAL_LOAD_ASYNC_TO_LDS_B128 (ASYNCcnt)
//   K2: dotsT[e][d] = (1/4096) * sum_n k[n][d] v[n][e]  per (b,h)
//   K3: out[b,n,h*64+e] = sum_d q[n][d] dotsT[e][d]  (fp32 output, async-staged)
// ---------------------------------------------------------------------------

typedef __bf16 bf16;
typedef bf16  v16bf __attribute__((ext_vector_type(16)));
typedef float v8f   __attribute__((ext_vector_type(8)));
typedef unsigned int u32x2 __attribute__((ext_vector_type(2)));
typedef unsigned int u32x4 __attribute__((ext_vector_type(4)));
typedef float        f32x4 __attribute__((ext_vector_type(4)));

union FragBF16 { u32x2 u2[4]; v16bf v; };   // 16 bf16 = 8 VGPRs per lane

__device__ __forceinline__ unsigned short f2bf(float f) {
    unsigned int u = __float_as_uint(f);
    u = u + 0x7FFFu + ((u >> 16) & 1u);   // RNE
    return (unsigned short)(u >> 16);
}

// 4x f32 -> 4x bf16 (packed), hardware cvt when available.
__device__ __forceinline__ u32x2 pack4_bf16(f32x4 g) {
    u32x2 p;
#if __has_builtin(__builtin_amdgcn_cvt_pk_bf16_f32)
    typedef __bf16 bf16x2 __attribute__((ext_vector_type(2)));
    bf16x2 lo = __builtin_amdgcn_cvt_pk_bf16_f32(g.x, g.y);
    bf16x2 hi = __builtin_amdgcn_cvt_pk_bf16_f32(g.z, g.w);
    p.x = __builtin_bit_cast(unsigned int, lo);
    p.y = __builtin_bit_cast(unsigned int, hi);
#else
    p.x = (unsigned)f2bf(g.x) | ((unsigned)f2bf(g.y) << 16);
    p.y = (unsigned)f2bf(g.z) | ((unsigned)f2bf(g.w) << 16);
#endif
    return p;
}

__device__ __forceinline__ v8f wmma_bf16(const v16bf& a, const v16bf& b, const v8f& c) {
    return __builtin_amdgcn_wmma_f32_16x16x32_bf16(false, a, false, b,
                                                   (short)0, c, false, false);
}

// Butterfly sum over each 16-lane half (xor masks < 16 never cross halves).
// ds_swizzle group-of-32 pattern: bits[14:10]=xor, [9:5]=or, [4:0]=and.
__device__ __forceinline__ float halfwave_sum16(float s) {
    s += __int_as_float(__builtin_amdgcn_ds_swizzle(__float_as_int(s), 0x041f)); // xor 1
    s += __int_as_float(__builtin_amdgcn_ds_swizzle(__float_as_int(s), 0x081f)); // xor 2
    s += __int_as_float(__builtin_amdgcn_ds_swizzle(__float_as_int(s), 0x101f)); // xor 4
    s += __int_as_float(__builtin_amdgcn_ds_swizzle(__float_as_int(s), 0x201f)); // xor 8
    return s;
}

// Async DMA: LDS[lds_addr] = MEM[gp .. gp+15], tracked by ASYNCcnt.
// Low 32 bits of a generic LDS pointer are the LDS byte address (ISA 10.2).
__device__ __forceinline__ void async_b128(const void* gp, const void* lp) {
    unsigned lds = (unsigned)(size_t)lp;
    asm volatile("global_load_async_to_lds_b128 %0, %1, off"
                 :: "v"(lds), "v"(gp)
                 : "memory");
}
__device__ __forceinline__ void wait_async0() {
    asm volatile("s_wait_asynccnt 0x0" ::: "memory");
}

// ---------------------------------------------------------------------------
// Kernel 0: one-time weight conversion  wT[n][k] = bf16(w[k][n]).
// grid (6, 512): coalesced fp32 reads along n, b16 scatter writes (L2-local).
// ---------------------------------------------------------------------------
__global__ __launch_bounds__(256) void w_to_bf16t(const float* __restrict__ w,
                                                  unsigned short* __restrict__ wT)
{
    const int n = blockIdx.x * 256 + threadIdx.x;   // 0..1535
    const int k = blockIdx.y;                        // 0..511
    wT[(size_t)n * 512 + k] = f2bf(w[(size_t)k * 1536 + n]);
}

// ---------------------------------------------------------------------------
// Kernel 1: qkv GEMM (M=32768, N=1536, K=512) with fused LayerNorm epilogue.
// Block: 128x128 tile, 256 thr (8 waves: 4 over M x 2 over N). K-step 32.
// ---------------------------------------------------------------------------
#define LDA 40   // LDS row stride (shorts) for 32-wide K tiles, pads banks

__device__ __forceinline__ void k1_stage(
    const float* __restrict__ x, const unsigned short* __restrict__ wT,
    int rowBase, int colBase, int kk, int tid,
    unsigned short* As, unsigned short* Bs)
{
    // phase 1: issue everything (4 fp32 loads for A, 2 async DMAs for B)
    f32x4 ga[4];
    #pragma unroll
    for (int i = 0; i < 4; ++i) {
        int f  = tid + 256 * i;            // 1024 float4 chunks
        int r  = f >> 3;
        int cv = (f & 7) * 4;
        const float* gp = &x[(size_t)(rowBase + r) * 512 + kk + cv];
        __builtin_prefetch(gp + 64, 0, 3);            // K tile after next
        ga[i] = *(const f32x4*)gp;
    }
    #pragma unroll
    for (int i = 0; i < 2; ++i) {          // 512 x 16B chunks, 2 per thread
        int f  = tid + 256 * i;
        int n  = f >> 2;                   // 128 rows, 4 chunks each
        int k0 = (f & 3) * 8;
        async_b128(&wT[(size_t)(colBase + n) * 512 + kk + k0],
                   &Bs[n * LDA + k0]);
    }
    // phase 2: convert + store A (waits are staggered by the scheduler)
    #pragma unroll
    for (int i = 0; i < 4; ++i) {
        int f  = tid + 256 * i;
        int r  = f >> 3;
        int cv = (f & 7) * 4;
        *(u32x2*)&As[r * LDA + cv] = pack4_bf16(ga[i]);
    }
}

__global__ __launch_bounds__(256) void qkv_gemm_ln(
    const float* __restrict__ x, const unsigned short* __restrict__ wT,
    const float* __restrict__ qg, const float* __restrict__ qb,
    const float* __restrict__ kg, const float* __restrict__ kb,
    unsigned short* __restrict__ q_ws, unsigned short* __restrict__ k_ws,
    unsigned short* __restrict__ v_ws)
{
    __shared__ alignas(16) unsigned short As[2][128 * LDA];
    __shared__ alignas(16) unsigned short Bs[2][128 * LDA];

    const int tid  = threadIdx.x;
    const int lane = tid & 31;
    const int wv   = tid >> 5;
    const int wm   = wv & 3;        // wave M slot (32 rows)
    const int wn   = wv >> 2;       // wave N slot (64 cols)
    const int rowBase = blockIdx.y * 128;
    const int colBase = blockIdx.x * 128;

    v8f acc[2][4] = {};

    // per-lane fragment geometry (K contiguous per lane half, same order for A and B)
    const int aM   = lane & 15;            // A: M = lane%16
    const int aKlo = (lane >> 4) * 16;     // A: K 0..15 / 16..31
    const int bN   = lane & 15;            // B: N = lane%16
    const int bKlo = (lane >> 4) * 16;     // B: K 0..15 / 16..31

    k1_stage(x, wT, rowBase, colBase, 0, tid, &As[0][0], &Bs[0][0]);
    int cur = 0;

    #pragma unroll 2
    for (int kk = 0; kk < 512; kk += 32) {
        wait_async0();                         // our DMAs into this buffer done
        __syncthreads();                       // everyone's writes visible
        if (kk + 32 < 512)
            k1_stage(x, wT, rowBase, colBase, kk + 32, tid,
                     &As[cur ^ 1][0], &Bs[cur ^ 1][0]);

        FragBF16 a[2], b[4];
        #pragma unroll
        for (int mt = 0; mt < 2; ++mt) {
            const unsigned short* p = &As[cur][(wm * 32 + mt * 16 + aM) * LDA + aKlo];
            a[mt].u2[0] = *(const u32x2*)(p);
            a[mt].u2[1] = *(const u32x2*)(p + 4);
            a[mt].u2[2] = *(const u32x2*)(p + 8);
            a[mt].u2[3] = *(const u32x2*)(p + 12);
        }
        #pragma unroll
        for (int nt = 0; nt < 4; ++nt) {
            const unsigned short* p = &Bs[cur][(wn * 64 + nt * 16 + bN) * LDA + bKlo];
            b[nt].u2[0] = *(const u32x2*)(p);
            b[nt].u2[1] = *(const u32x2*)(p + 4);
            b[nt].u2[2] = *(const u32x2*)(p + 8);
            b[nt].u2[3] = *(const u32x2*)(p + 12);
        }
        #pragma unroll
        for (int mt = 0; mt < 2; ++mt)
            #pragma unroll
            for (int nt = 0; nt < 4; ++nt)
                acc[mt][nt] = wmma_bf16(a[mt].v, b[nt].v, acc[mt][nt]);
        cur ^= 1;
    }

    // --- epilogue: fused LayerNorm (q,k) + scatter into head-major bf16 ---
    const int col0  = colBase + wn * 64;       // aligned to one head's 64-chunk
    const int which = col0 >> 9;               // 0=q 1=k 2=v
    const int head  = (col0 & 511) >> 6;
    const float* gam = (which == 0) ? qg : kg;
    const float* bet = (which == 0) ? qb : kb;
    unsigned short* dst = (which == 0) ? q_ws : (which == 1) ? k_ws : v_ws;
    const int laneN = lane & 15;
    const int hiM   = (lane >> 4) * 8;         // C-frag: lanes 16..31 hold M=r+8

    float gamv[4], betv[4];
    #pragma unroll
    for (int nt = 0; nt < 4; ++nt) {           // hoisted, uniform per lane
        gamv[nt] = gam[nt * 16 + laneN];
        betv[nt] = bet[nt * 16 + laneN];
    }

    #pragma unroll
    for (int mt = 0; mt < 2; ++mt) {
        #pragma unroll
        for (int r = 0; r < 8; ++r) {
            float vals[4], s = 0.f, sq = 0.f;
            #pragma unroll
            for (int nt = 0; nt < 4; ++nt) {
                float t = acc[mt][nt][r];
                vals[nt] = t; s += t; sq += t * t;
            }
            // 16-lane allreduce keeps M=r (lanes 0-15) and M=r+8 (16-31) apart
            s  = halfwave_sum16(s);
            sq = halfwave_sum16(sq);
            const float mean = s * 0.015625f;
            const float var  = sq * 0.015625f - mean * mean;
            const float rstd = rsqrtf(var + 1e-5f);
            const int row  = rowBase + wm * 32 + mt * 16 + r + hiM;
            const int bIdx = row >> 12;
            const int nIdx = row & 4095;
            const size_t base = ((size_t)(bIdx * 8 + head) * 4096 + nIdx) * 64;
            #pragma unroll
            for (int nt = 0; nt < 4; ++nt) {
                const int d = nt * 16 + laneN;
                float o = vals[nt];
                if (which < 2) o = (o - mean) * rstd * gamv[nt] + betv[nt];
                dst[base + d] = f2bf(o);
            }
        }
    }
}

// ---------------------------------------------------------------------------
// Kernel 2: dotsT = (1/4096) * (k^T v) per (b,h). One block per (b,h).
// 16 output 16x16 tiles split 2-per-wave; LDS tiles transposed-on-store so
// both A (k^T) and B (v) fragments read contiguously along K=n.
// ---------------------------------------------------------------------------
#define LDT 36

__device__ __forceinline__ void k2_stage(
    const unsigned short* __restrict__ kp, const unsigned short* __restrict__ vp,
    int nb, int lrow, int lseg, unsigned short* kT, unsigned short* vT)
{
    union { u32x4 u; unsigned short s[8]; } gk, gv;
    gk.u = *(const u32x4*)&kp[(size_t)(nb + lrow) * 64 + lseg];
    gv.u = *(const u32x4*)&vp[(size_t)(nb + lrow) * 64 + lseg];
    #pragma unroll
    for (int e = 0; e < 8; ++e) {
        kT[(lseg + e) * LDT + lrow] = gk.s[e];
        vT[(lseg + e) * LDT + lrow] = gv.s[e];
    }
}

__global__ __launch_bounds__(256) void ktv_dots(
    const unsigned short* __restrict__ k_ws,
    const unsigned short* __restrict__ v_ws,
    unsigned short* __restrict__ dots_ws)
{
    __shared__ alignas(16) unsigned short kT[2][64 * LDT];   // [d][n]
    __shared__ alignas(16) unsigned short vT[2][64 * LDT];   // [e][n]

    const int tid = threadIdx.x, lane = tid & 31, wv = tid >> 5;
    const int bh  = blockIdx.x;
    const unsigned short* kp = k_ws + (size_t)bh * 4096 * 64;
    const unsigned short* vp = v_ws + (size_t)bh * 4096 * 64;

    const int myMt = wv & 3;            // d tile
    const int myNt = (wv >> 2) * 2;     // first of two e tiles
    v8f acc[2] = {};

    const int aM = lane & 15, aKlo = (lane >> 4) * 16;
    const int bN = lane & 15, bKlo = (lane >> 4) * 16;
    const int lrow = tid >> 3;          // 0..31 (n within tile)
    const int lseg = (tid & 7) * 8;     // 8-element column segment

    k2_stage(kp, vp, 0, lrow, lseg, &kT[0][0], &vT[0][0]);
    int cur = 0;

    #pragma unroll 2
    for (int nb = 0; nb < 4096; nb += 32) {
        __syncthreads();
        if (nb + 32 < 4096)
            k2_stage(kp, vp, nb + 32, lrow, lseg, &kT[cur ^ 1][0], &vT[cur ^ 1][0]);

        FragBF16 a, b0, b1;
        {
            const unsigned short* p = &kT[cur][(myMt * 16 + aM) * LDT + aKlo];
            a.u2[0] = *(const u32x2*)(p);      a.u2[1] = *(const u32x2*)(p + 4);
            a.u2[2] = *(const u32x2*)(p + 8); a.u2[3] = *(const u32x2*)(p + 12);
        }
        {
            const unsigned short* p = &vT[cur][(myNt * 16 + bN) * LDT + bKlo];
            b0.u2[0] = *(const u32x2*)(p);     b0.u2[1] = *(const u32x2*)(p + 4);
            b0.u2[2] = *(const u32x2*)(p + 8); b0.u2[3] = *(const u32x2*)(p + 12);
        }
        {
            const unsigned short* p = &vT[cur][((myNt + 1) * 16 + bN) * LDT + bKlo];
            b1.u2[0] = *(const u32x2*)(p);     b1.u2[1] = *(const u32x2*)(p + 4);
            b1.u2[2] = *(const u32x2*)(p + 8); b1.u2[3] = *(const u32x2*)(p + 12);
        }
        acc[0] = wmma_bf16(a.v, b0.v, acc[0]);
        acc[1] = wmma_bf16(a.v, b1.v, acc[1]);
        cur ^= 1;
    }

    unsigned short* dp = dots_ws + (size_t)bh * 4096;   // dotsT [e][d]
    const int hiM = (lane >> 4) * 8;
    #pragma unroll
    for (int j = 0; j < 2; ++j) {
        #pragma unroll
        for (int r = 0; r < 8; ++r) {
            const int d = myMt * 16 + r + hiM;
            const int e = (myNt + j) * 16 + (lane & 15);
            dp[e * 64 + d] = f2bf(acc[j][r] * (1.0f / 4096.0f));
        }
    }
}

// ---------------------------------------------------------------------------
// Kernel 3: out = q @ dots, output fp32 in [B, L*W, H*64] layout.
// q tile (16KB) + dots tile (8KB) staged via async DMA to LDS.
// ---------------------------------------------------------------------------
#define LDQ 72
__global__ __launch_bounds__(256) void q_dots_out(
    const unsigned short* __restrict__ q_ws,
    const unsigned short* __restrict__ dots_ws,
    float* __restrict__ out)
{
    __shared__ alignas(16) unsigned short qS[128 * LDQ];  // [row][d]
    __shared__ alignas(16) unsigned short dS[64 * LDQ];   // dotsT [e][d]

    const int tid = threadIdx.x, lane = tid & 31, wv = tid >> 5;
    const int mblk = blockIdx.x;
    const int bh   = blockIdx.y;
    const unsigned short* qp = q_ws + (size_t)bh * 4096 * 64 + (size_t)mblk * 128 * 64;
    const unsigned short* dp = dots_ws + (size_t)bh * 4096;

    // async-stage q tile: 128x64 shorts = 1024 x 16B, 4 per thread
    #pragma unroll
    for (int i = 0; i < 4; ++i) {
        int u = tid + 256 * i;
        int r = u >> 3, seg = (u & 7) * 8;
        async_b128(&qp[(size_t)r * 64 + seg], &qS[r * LDQ + seg]);
    }
    // async-stage dotsT tile: 64x64 shorts = 512 x 16B, 2 per thread
    #pragma unroll
    for (int i = 0; i < 2; ++i) {
        int u = tid + 256 * i;
        int r = u >> 3, seg = (u & 7) * 8;
        async_b128(&dp[(size_t)r * 64 + seg], &dS[r * LDQ + seg]);
    }
    wait_async0();
    __syncthreads();

    const int wm = wv & 3, wn = wv >> 2;
    v8f acc[2][2] = {};
    const int aM = lane & 15, aKlo = (lane >> 4) * 16;
    const int bN = lane & 15, bKlo = (lane >> 4) * 16;

    #pragma unroll
    for (int kk = 0; kk < 64; kk += 32) {
        FragBF16 a[2], b[2];
        #pragma unroll
        for (int mt = 0; mt < 2; ++mt) {
            const unsigned short* p = &qS[(wm * 32 + mt * 16 + aM) * LDQ + kk + aKlo];
            a[mt].u2[0] = *(const u32x2*)(p);      a[mt].u2[1] = *(const u32x2*)(p + 4);
            a[mt].u2[2] = *(const u32x2*)(p + 8); a[mt].u2[3] = *(const u32x2*)(p + 12);
        }
        #pragma unroll
        for (int nt = 0; nt < 2; ++nt) {
            const unsigned short* p = &dS[(wn * 32 + nt * 16 + bN) * LDQ + kk + bKlo];
            b[nt].u2[0] = *(const u32x2*)(p);     b[nt].u2[1] = *(const u32x2*)(p + 4);
            b[nt].u2[2] = *(const u32x2*)(p + 8); b[nt].u2[3] = *(const u32x2*)(p + 12);
        }
        #pragma unroll
        for (int mt = 0; mt < 2; ++mt)
            #pragma unroll
            for (int nt = 0; nt < 2; ++nt)
                acc[mt][nt] = wmma_bf16(a[mt].v, b[nt].v, acc[mt][nt]);
    }

    const int bIdx = bh >> 3, head = bh & 7;
    const int hiM = (lane >> 4) * 8;
    #pragma unroll
    for (int mt = 0; mt < 2; ++mt)
        #pragma unroll
        for (int nt = 0; nt < 2; ++nt)
            #pragma unroll
            for (int r = 0; r < 8; ++r) {
                const int row = mblk * 128 + wm * 32 + mt * 16 + r + hiM;
                const int e   = wn * 32 + nt * 16 + (lane & 15);
                out[((size_t)(bIdx * 4096 + row)) * 512 + head * 64 + e] = acc[mt][nt][r];
            }
}

// ---------------------------------------------------------------------------
extern "C" void kernel_launch(void* const* d_in, const int* in_sizes, int n_in,
                              void* d_out, int out_size, void* d_ws, size_t ws_size,
                              hipStream_t stream) {
    const float* x  = (const float*)d_in[0];
    const float* w  = (const float*)d_in[1];
    const float* qg = (const float*)d_in[2];
    const float* qb = (const float*)d_in[3];
    const float* kg = (const float*)d_in[4];
    const float* kb = (const float*)d_in[5];
    float* out = (float*)d_out;

    const size_t headElems = (size_t)8 * 8 * 4096 * 64;   // 16,777,216 bf16
    unsigned short* q_ws    = (unsigned short*)d_ws;
    unsigned short* k_ws    = q_ws + headElems;
    unsigned short* v_ws    = k_ws + headElems;
    unsigned short* dots_ws = v_ws + headElems;           // 64*64*64 shorts
    unsigned short* wT_ws   = dots_ws + (size_t)64 * 64 * 64;  // 1536*512 shorts
                                                          // total ~98 MB

    w_to_bf16t<<<dim3(6, 512), 256, 0, stream>>>(w, wT_ws);
    qkv_gemm_ln<<<dim3(12, 256), 256, 0, stream>>>(x, wT_ws, qg, qb, kg, kb,
                                                   q_ws, k_ws, v_ws);
    ktv_dots<<<dim3(64), 256, 0, stream>>>(k_ws, v_ws, dots_ws);
    q_dots_out<<<dim3(32, 64), 256, 0, stream>>>(q_ws, dots_ws, out);
}
